// SelfAttention_15144054686524
// MI455X (gfx1250) — compile-verified, matrix operand order checked by reading, not run on hardware
//
#include <hip/hip_runtime.h>
#include <hip/hip_bf16.h>

#define NUM_HEADS 16
#define DIM 2048
#define SEQ 2048
#define DH 128
#define EPS 1.1920929e-07f

typedef __bf16 bf16_t;
typedef __attribute__((ext_vector_type(16))) __bf16 v16bf;
typedef __attribute__((ext_vector_type(8)))  __bf16 v8bf;
typedef __attribute__((ext_vector_type(8)))  float  v8f;

#define WMMA_BF16(a, b, c) \
  __builtin_amdgcn_wmma_f32_16x16x32_bf16(false, (a), false, (b), (short)0, (c), false, false)

#define KV_TILE   (32 * 128)   // attention K/V staging tile (8KB bf16)
#define AT_ELEMS  (256 * 32)   // GEMM A stage tile (16KB)
#define BT_ELEMS  (128 * 32)   // GEMM B stage tile (8KB)

// Load a 16-element bf16 A/B fragment for v_wmma_f32_16x16x32_bf16.
// Caller passes p = row_base + k0 + ((lane>=16)?8:0).
// elements 0..7 = [p..p+7], elements 8..15 = [p+16..p+23].
__device__ __forceinline__ v16bf load_frag16(const bf16_t* p) {
  union { v16bf v; v8bf h[2]; } u;
  u.h[0] = *reinterpret_cast<const v8bf*>(p);
  u.h[1] = *reinterpret_cast<const v8bf*>(p + 16);
  return u.v;
}

// Async global -> LDS copy, 16B per lane, tracked by ASYNCcnt.
__device__ __forceinline__ void async_g2l_b128(bf16_t* lptr, const bf16_t* gptr) {
  unsigned int lo = (unsigned int)(uintptr_t)lptr;  // low 32 bits = LDS byte offset
  asm volatile("global_load_async_to_lds_b128 %0, %1, off"
               :: "v"(lo), "v"(gptr) : "memory");
}
__device__ __forceinline__ void wait_async_le6() {
  asm volatile("s_wait_asynccnt 0x6" ::: "memory");
}
__device__ __forceinline__ void wait_async_le4() {
  asm volatile("s_wait_asynccnt 0x4" ::: "memory");
}
__device__ __forceinline__ void wait_async_0() {
  asm volatile("s_wait_asynccnt 0x0" ::: "memory");
}

// ---------------------------------------------------------------- convert ---
__global__ __launch_bounds__(256) void cvt_f32_bf16(const float* __restrict__ s,
                                                    bf16_t* __restrict__ d, int n) {
  int i = blockIdx.x * blockDim.x + threadIdx.x;
  if (i < n) d[i] = (bf16_t)s[i];
}

// ------------------------------------------ cooperative 256x128 GEMM core ---
// Block (8 waves) computes C[256,128] = A[256,K] * B[128,K]^T with LDS
// double-buffered async staging. Each wave owns 32 rows x 128 cols in
// acc[2][8]; every B fragment feeds two WMMAs.
__device__ __forceinline__ void gemm_core_256x128(
    const bf16_t* __restrict__ Ag,  // global, row 0 = block's m0, stride DIM
    const bf16_t* __restrict__ Bg,  // global, row 0 = block's n0, stride DIM
    bf16_t* __restrict__ As,        // shared [2*AT_ELEMS]
    bf16_t* __restrict__ Bs,        // shared [2*BT_ELEMS]
    v8f acc[2][8])
{
  const int tid   = threadIdx.x;
  const int lane  = tid & 31;
  const int w     = tid >> 5;
  const int lr    = lane & 15;
  const int khalf = (lane >> 4) << 3;
  // copy indices: 16B chunks; A tile 1024 chunks (4/thread), B tile 512 (2/thread)
  const int cr  = tid >> 2;        // row 0..63 (+64/128/192)
  const int cc8 = (tid & 3) * 8;   // col offset in elements

#pragma unroll
  for (int s2 = 0; s2 < 2; s2++)
#pragma unroll
    for (int n = 0; n < 8; n++) acc[s2][n] = {};

  // prologue: stage 0 into buffer 0
#pragma unroll
  for (int j = 0; j < 4; j++)
    async_g2l_b128(As + (cr + j * 64) * 32 + cc8, Ag + (size_t)(cr + j * 64) * DIM + cc8);
#pragma unroll
  for (int j = 0; j < 2; j++)
    async_g2l_b128(Bs + (cr + j * 64) * 32 + cc8, Bg + (size_t)(cr + j * 64) * DIM + cc8);

  const int NS = DIM / 32;
  for (int s = 0; s < NS; s++) {
    const int cur = s & 1;
    if (s + 1 < NS) {  // stage s+1 into the other buffer, then wait for stage s
      const int kk = (s + 1) * 32;
      bf16_t* An = As + (cur ^ 1) * AT_ELEMS;
      bf16_t* Bn = Bs + (cur ^ 1) * BT_ELEMS;
#pragma unroll
      for (int j = 0; j < 4; j++)
        async_g2l_b128(An + (cr + j * 64) * 32 + cc8, Ag + (size_t)(cr + j * 64) * DIM + kk + cc8);
#pragma unroll
      for (int j = 0; j < 2; j++)
        async_g2l_b128(Bn + (cr + j * 64) * 32 + cc8, Bg + (size_t)(cr + j * 64) * DIM + kk + cc8);
      wait_async_le6();
    } else {
      wait_async_0();
    }
    __syncthreads();

    const bf16_t* At = As + cur * AT_ELEMS + (w * 32 + lr) * 32 + khalf;
    const bf16_t* Bt = Bs + cur * BT_ELEMS;
    v16bf a0 = load_frag16(At);
    v16bf a1 = load_frag16(At + 16 * 32);
#pragma unroll
    for (int n = 0; n < 8; n++) {
      v16bf b = load_frag16(Bt + (n * 16 + lr) * 32 + khalf);
      acc[0][n] = WMMA_BF16(a0, b, acc[0][n]);
      acc[1][n] = WMMA_BF16(a1, b, acc[1][n]);
    }
    __syncthreads();
  }
}

// -------------------------------------------- fused Q/K GEMM + RMS + RoPE ---
__global__ __launch_bounds__(256) void gemm_qk_norm_rope(
    const bf16_t* __restrict__ xb,  // [T, D]
    const bf16_t* __restrict__ wb,  // [D, D]
    bf16_t* __restrict__ outh)      // [H][T][DH]
{
  __shared__ __align__(128) bf16_t As[2 * AT_ELEMS];
  __shared__ __align__(128) bf16_t Bs[2 * BT_ELEMS];
  const int lane = threadIdx.x & 31;
  const int w    = threadIdx.x >> 5;
  const int mblk = blockIdx.x >> 4;  // 0..7 (256 rows each)
  const int h    = blockIdx.x & 15;
  const int lr   = lane & 15;
  const int hi8  = (lane >> 4) << 3;

  v8f acc[2][8];
  gemm_core_256x128(xb + (size_t)(mblk * 256) * DIM, wb + (size_t)(h * DH) * DIM,
                    As, Bs, acc);

  const float kLog = 9.210340371976184f / 64.0f;  // ln(10000)/64
#pragma unroll
  for (int sub = 0; sub < 2; sub++) {
    const int tb = mblk * 256 + w * 32 + sub * 16;
    // RMS-norm over the 128 columns of each row (reduce across 16-lane group).
#pragma unroll
    for (int r = 0; r < 8; r++) {
      float s = 0.f;
#pragma unroll
      for (int n = 0; n < 8; n++) s += acc[sub][n][r] * acc[sub][n][r];
#pragma unroll
      for (int m = 1; m <= 8; m <<= 1) s += __shfl_xor(s, m, 32);
      float inv = rsqrtf(s * (1.0f / 128.0f) + EPS);
#pragma unroll
      for (int n = 0; n < 8; n++) acc[sub][n][r] *= inv;
    }
    // RoPE: pair (d, d+64) -> (acc[n], acc[n+4]) same lane. inv_freq = B^(-j/64).
#pragma unroll
    for (int r = 0; r < 8; r++) {
      float t = (float)(tb + r + hi8);
#pragma unroll
      for (int n = 0; n < 4; n++) {
        int j = n * 16 + lr;
        float ang = t * __expf(-(float)j * kLog);
        float cv = __cosf(ang), sv = __sinf(ang);
        float x1 = acc[sub][n][r], x2 = acc[sub][n + 4][r];
        acc[sub][n][r]     =  x1 * cv + x2 * sv;
        acc[sub][n + 4][r] = -x1 * sv + x2 * cv;
      }
    }
#pragma unroll
    for (int r = 0; r < 8; r++) {
      int t = tb + r + hi8;
#pragma unroll
      for (int n = 0; n < 8; n++)
        outh[((size_t)h * SEQ + t) * DH + n * 16 + lr] = (bf16_t)acc[sub][n][r];
    }
  }
}

// ------------------------------------- V GEMM + lambda blend (transposed) ---
__global__ __launch_bounds__(256) void gemm_v_blend(
    const bf16_t* __restrict__ xb, const bf16_t* __restrict__ wb,
    const float* __restrict__ vi, const float* __restrict__ lambdas,
    bf16_t* __restrict__ vt)        // [H][DH][T]
{
  __shared__ __align__(128) bf16_t As[2 * AT_ELEMS];
  __shared__ __align__(128) bf16_t Bs[2 * BT_ELEMS];
  const int lane = threadIdx.x & 31;
  const int w    = threadIdx.x >> 5;
  const int mblk = blockIdx.x >> 4;
  const int h    = blockIdx.x & 15;
  const int lr   = lane & 15;
  const int hi8  = (lane >> 4) << 3;
  const float l0 = lambdas[0], l1 = lambdas[1];

  v8f acc[2][8];
  gemm_core_256x128(xb + (size_t)(mblk * 256) * DIM, wb + (size_t)(h * DH) * DIM,
                    As, Bs, acc);

#pragma unroll
  for (int sub = 0; sub < 2; sub++) {
#pragma unroll
    for (int r = 0; r < 8; r++) {
      int t = mblk * 256 + w * 32 + sub * 16 + r + hi8;
#pragma unroll
      for (int n = 0; n < 8; n++) {
        int d = n * 16 + lr;
        float v = l0 * acc[sub][n][r] + l1 * vi[(size_t)t * DIM + h * DH + d];
        vt[((size_t)h * DH + d) * SEQ + t] = (bf16_t)v;
      }
    }
  }
}

// ------------------------------------------------- flash causal attention ---
// Block (8 waves) owns 128 queries of one head; K/V 32-key blocks staged in
// LDS (double-buffered async) and shared by all 8 waves.
__global__ __launch_bounds__(256) void attn_flash(
    const bf16_t* __restrict__ qh, const bf16_t* __restrict__ kh,
    const bf16_t* __restrict__ vt, bf16_t* __restrict__ yb)
{
  __shared__ __align__(128) bf16_t Ks[2 * KV_TILE];  // [32 keys][128 d]
  __shared__ __align__(128) bf16_t Vs[2 * KV_TILE];  // [128 d][32 keys]
  __shared__ __align__(16)  bf16_t plds[8][16 * 32];

  const int tid   = threadIdx.x;
  const int lane  = tid & 31;
  const int w     = tid >> 5;
  const int h     = blockIdx.x >> 4;
  const int qb    = blockIdx.x & 15;
  const int q0    = qb * 128;
  const int lr    = lane & 15;
  const int khalf = (lane >> 4) << 3;
  const int hi8   = (lane >> 4) << 3;

  const bf16_t* Q = qh + (size_t)h * SEQ * DH;
  const bf16_t* K = kh + (size_t)h * SEQ * DH;
  const bf16_t* V = vt + (size_t)h * DH * SEQ;

  // copy indices: K tile 32x128 (16 chunks/row), V tile 128x32 (4 chunks/row)
  const int krow = tid >> 4, kcc = (tid & 15) * 8;
  const int vrow = tid >> 2, vcc = (tid & 3) * 8;

  v16bf qfrag[4];
#pragma unroll
  for (int kk = 0; kk < 4; kk++)
    qfrag[kk] = load_frag16(Q + (size_t)(q0 + w * 16 + lr) * DH + kk * 32 + khalf);

  v8f o[8];
#pragma unroll
  for (int n = 0; n < 8; n++) o[n] = {};
  float mrow[8], lrow[8];
#pragma unroll
  for (int r = 0; r < 8; r++) { mrow[r] = -1e30f; lrow[r] = 0.f; }

  const float sc = 0.08838834764831845f;  // 1/sqrt(128)
  const int NS = q0 / 32 + 4;             // key blocks needed by this query block
  const int kmax_w = q0 + w * 16 + 15;    // this wave's causal limit

  // prologue: stage key-block 0
  {
    async_g2l_b128(Ks + krow * 128 + kcc,        K + (size_t)krow * DH + kcc);
    async_g2l_b128(Ks + (krow + 16) * 128 + kcc, K + (size_t)(krow + 16) * DH + kcc);
    async_g2l_b128(Vs + vrow * 32 + vcc,         V + (size_t)vrow * SEQ + vcc);
    async_g2l_b128(Vs + (vrow + 64) * 32 + vcc,  V + (size_t)(vrow + 64) * SEQ + vcc);
  }

  for (int kb = 0; kb < NS; kb++) {
    const int cur = kb & 1;
    if (kb + 1 < NS) {
      const int t0 = (kb + 1) * 32;
      bf16_t* Kn = Ks + (cur ^ 1) * KV_TILE;
      bf16_t* Vn = Vs + (cur ^ 1) * KV_TILE;
      async_g2l_b128(Kn + krow * 128 + kcc,        K + (size_t)(t0 + krow) * DH + kcc);
      async_g2l_b128(Kn + (krow + 16) * 128 + kcc, K + (size_t)(t0 + krow + 16) * DH + kcc);
      async_g2l_b128(Vn + vrow * 32 + vcc,         V + (size_t)vrow * SEQ + t0 + vcc);
      async_g2l_b128(Vn + (vrow + 64) * 32 + vcc,  V + (size_t)(vrow + 64) * SEQ + t0 + vcc);
      wait_async_le4();
    } else {
      wait_async_0();
    }
    __syncthreads();

    if (kb * 32 <= kmax_w) {
      const bf16_t* Kt = Ks + cur * KV_TILE;
      const bf16_t* Vt = Vs + cur * KV_TILE;

      // S = Q * K^T (two 16x16 tiles over 32 keys)
      v16bf kf0[4], kf1[4];
#pragma unroll
      for (int kk = 0; kk < 4; kk++) {
        kf0[kk] = load_frag16(Kt + (size_t)lr * DH + kk * 32 + khalf);
        kf1[kk] = load_frag16(Kt + (size_t)(16 + lr) * DH + kk * 32 + khalf);
      }
      v8f s0 = {}, s1 = {};
#pragma unroll
      for (int kk = 0; kk < 4; kk++) {
        s0 = WMMA_BF16(qfrag[kk], kf0[kk], s0);
        s1 = WMMA_BF16(qfrag[kk], kf1[kk], s1);
      }

      const int col = kb * 32 + lr;
#pragma unroll
      for (int r = 0; r < 8; r++) {
        int qrow = q0 + w * 16 + r + hi8;
        float v0 = (col      <= qrow) ? s0[r] * sc : -1e30f;
        float v1 = (col + 16 <= qrow) ? s1[r] * sc : -1e30f;
        float mx = fmaxf(v0, v1);
#pragma unroll
        for (int m = 1; m <= 8; m <<= 1) mx = fmaxf(mx, __shfl_xor(mx, m, 32));
        float mnew  = fmaxf(mrow[r], mx);
        float alpha = __expf(mrow[r] - mnew);
        float p0 = __expf(v0 - mnew), p1 = __expf(v1 - mnew);
        s0[r] = p0; s1[r] = p1;
        float rs = p0 + p1;
#pragma unroll
        for (int m = 1; m <= 8; m <<= 1) rs += __shfl_xor(rs, m, 32);
        lrow[r] = lrow[r] * alpha + rs;
        mrow[r] = mnew;
#pragma unroll
        for (int n = 0; n < 8; n++) o[n][r] *= alpha;
      }

      // P: C-layout -> A-layout via wave-private LDS tile
      bf16_t* pl = plds[w];
#pragma unroll
      for (int r = 0; r < 8; r++) {
        int prow = r + hi8;
        pl[prow * 32 + lr]      = (bf16_t)s0[r];
        pl[prow * 32 + 16 + lr] = (bf16_t)s1[r];
      }
      asm volatile("s_wait_dscnt 0x0" ::: "memory");
      v16bf pf = load_frag16(pl + lr * 32 + khalf);

      // O += P * V
      v16bf vf[8];
#pragma unroll
      for (int n = 0; n < 8; n++)
        vf[n] = load_frag16(Vt + (size_t)(n * 16 + lr) * 32 + khalf);
#pragma unroll
      for (int n = 0; n < 8; n++) o[n] = WMMA_BF16(pf, vf[n], o[n]);
    }
    __syncthreads();
  }

#pragma unroll
  for (int r = 0; r < 8; r++) {
    float inv = 1.0f / lrow[r];
    int t = q0 + w * 16 + r + hi8;
#pragma unroll
    for (int n = 0; n < 8; n++)
      yb[(size_t)t * DIM + h * DH + n * 16 + lr] = (bf16_t)(o[n][r] * inv);
  }
}

// -------------------------------------------------- final projection GEMM ---
__global__ __launch_bounds__(256) void gemm_proj(
    const bf16_t* __restrict__ A, const bf16_t* __restrict__ B,
    float* __restrict__ C)
{
  __shared__ __align__(128) bf16_t As[2 * AT_ELEMS];
  __shared__ __align__(128) bf16_t Bs[2 * BT_ELEMS];
  const int lane = threadIdx.x & 31;
  const int w    = threadIdx.x >> 5;
  const int mblk = blockIdx.x >> 4;  // 0..7
  const int nblk = blockIdx.x & 15;
  const int lr   = lane & 15;
  const int hi8  = (lane >> 4) << 3;

  v8f acc[2][8];
  gemm_core_256x128(A + (size_t)(mblk * 256) * DIM, B + (size_t)(nblk * 128) * DIM,
                    As, Bs, acc);

#pragma unroll
  for (int sub = 0; sub < 2; sub++) {
#pragma unroll
    for (int r = 0; r < 8; r++) {
      int m = mblk * 256 + w * 32 + sub * 16 + r + hi8;
#pragma unroll
      for (int n = 0; n < 8; n++)
        C[(size_t)m * DIM + nblk * 128 + n * 16 + lr] = acc[sub][n][r];
    }
  }
}

// ---------------------------------------------------------------- launch ----
extern "C" void kernel_launch(void* const* d_in, const int* in_sizes, int n_in,
                              void* d_out, int out_size, void* d_ws, size_t ws_size,
                              hipStream_t stream) {
  const float* x   = (const float*)d_in[0];
  const float* vi  = (const float*)d_in[1];
  const float* wq  = (const float*)d_in[2];
  const float* wk  = (const float*)d_in[3];
  const float* wv  = (const float*)d_in[4];
  const float* wp  = (const float*)d_in[5];
  const float* lam = (const float*)d_in[6];
  float* out = (float*)d_out;

  const size_t MAT  = (size_t)SEQ * DIM;      // 4M elems
  const size_t MATB = MAT * sizeof(bf16_t);   // 8 MB
  char* ws = (char*)d_ws;
  bf16_t* xb  = (bf16_t*)(ws + 0 * MATB);
  bf16_t* wqb = (bf16_t*)(ws + 1 * MATB);
  bf16_t* wkb = (bf16_t*)(ws + 2 * MATB);
  bf16_t* wvb = (bf16_t*)(ws + 3 * MATB);
  bf16_t* wpb = (bf16_t*)(ws + 4 * MATB);
  bf16_t* qh  = (bf16_t*)(ws + 5 * MATB);     // [H][T][DH]
  bf16_t* kh  = (bf16_t*)(ws + 6 * MATB);     // [H][T][DH]
  bf16_t* vt  = (bf16_t*)(ws + 7 * MATB);     // [H][DH][T]
  bf16_t* yb  = (bf16_t*)(ws + 8 * MATB);     // [T][D]

  const int cvtBlocks = (int)(MAT / 256);
  cvt_f32_bf16<<<cvtBlocks, 256, 0, stream>>>(x,  xb,  (int)MAT);
  cvt_f32_bf16<<<cvtBlocks, 256, 0, stream>>>(wq, wqb, (int)MAT);
  cvt_f32_bf16<<<cvtBlocks, 256, 0, stream>>>(wk, wkb, (int)MAT);
  cvt_f32_bf16<<<cvtBlocks, 256, 0, stream>>>(wv, wvb, (int)MAT);
  cvt_f32_bf16<<<cvtBlocks, 256, 0, stream>>>(wp, wpb, (int)MAT);

  // GEMMs: 128 blocks = 8 row-blocks (256 rows) x 16 heads / n-tiles
  gemm_qk_norm_rope<<<128, 256, 0, stream>>>(xb, wqb, qh);
  gemm_qk_norm_rope<<<128, 256, 0, stream>>>(xb, wkb, kh);
  gemm_v_blend<<<128, 256, 0, stream>>>(xb, wvb, vi, lam, vt);

  // attention: 256 blocks = 16 heads x 16 query-blocks (128 queries)
  attn_flash<<<256, 256, 0, stream>>>(qh, kh, vt, yb);

  gemm_proj<<<128, 256, 0, stream>>>(yb, wpb, out);
}
